// SwitchSAE_23124103922404
// MI455X (gfx1250) — matrix operand order repaired and checked by reading, not compiled, not run on hardware
//
#include <hip/hip_runtime.h>

// ---------------------------------------------------------------------------
// Switch-SAE top-1 routed encode/decode for MI455X (gfx1250, wave32, WMMA).
//   Kernel P: pre-pack f32 weights -> bf16 in per-lane WMMA B-fragment order.
//   Kernel 1: router (softmax/top-1) + bin tokens by expert.
//   Kernel 2: grouped fused GEMM: relu((x-b_pre)@enc_e) @ dec_e, scaled.
// Math path: bf16 WMMA (v_wmma_f32_16x16x32_bf16), f32 accumulate.
// ---------------------------------------------------------------------------

#define NTOK 8192
#define DIN  768
#define DK   3072
#define NE   8
#define MT   32          // tokens per block tile
#define CH   128         // latent columns per chunk (24 chunks)
#define PX   776         // sXc pitch in bf16 elems (388 dw, %8==4 -> conflict-free b128)
#define LP   136         // sL  pitch in bf16 elems  (68 dw, %8==4)

#define ENC_KT (DIN / 32)   // 24  K-tiles per enc column strip
#define ENC_NT (DK / 16)    // 192 N-tiles (latent cols)
#define DEC_KT (DK / 32)    // 96  K-tiles (latent rows)
#define DEC_NT (DIN / 16)   // 48  N-tiles (output cols)
#define TILE_ELEMS 512      // 32x16 bf16 tile = 512 elems = 1KB

typedef __attribute__((ext_vector_type(16))) __bf16          v16bf;
typedef __attribute__((ext_vector_type(8)))  float           v8f;
typedef __attribute__((ext_vector_type(16))) unsigned short  v16u;
typedef __attribute__((ext_vector_type(8)))  unsigned short  v8u;

__device__ __forceinline__ unsigned short f2bf(float f) {
    unsigned int u = __float_as_uint(f);
    unsigned int r = u + 0x7FFFu + ((u >> 16) & 1u);   // round-to-nearest-even
    return (unsigned short)(r >> 16);
}

__device__ __forceinline__ v8f wmma_bf16(v16bf a, v16bf b, v8f c) {
    return __builtin_amdgcn_wmma_f32_16x16x32_bf16(false, a, false, b,
                                                   (short)0, c, false, false);
}

// A fragment (16x32 bf16) from an LDS row pointer, in contiguous k order, matching B:
// lane m=l&15, g=l>>4; dwords 0..7 hold K = kk+g*16+{0..15} (contiguous).
__device__ __forceinline__ v16bf load_a_lds(const unsigned short* rowp, int kk, int g) {
    union { v8u h[2]; v16bf bf; } r;
    r.h[0] = *(const v8u*)(rowp + kk + g * 16);
    r.h[1] = *(const v8u*)(rowp + kk + g * 16 + 8);
    return r.bf;
}

// B fragment (32x16) straight from the pre-packed bf16 buffer: the packer laid
// out each tile so lane l's 16 values are contiguous -> two b128 loads, no VALU.
__device__ __forceinline__ v16bf load_b_packed(const unsigned short* tp, int lane) {
    union { v8u h[2]; v16bf bf; } r;
    const v8u* p = (const v8u*)(tp + lane * 16);
    r.h[0] = p[0];
    r.h[1] = p[1];
    return r.bf;
}

// Fallback: B fragment packed on the fly from f32 global weights.
// Per-lane: n=l&15, g=l>>4; dword j holds {B[g*16+2j][n], B[g*16+2j+1][n]}.
__device__ __forceinline__ v16bf load_b_f32(const float* __restrict__ bp, int ldb) {
    union { v16u u; v16bf bf; } r;
#pragma unroll
    for (int j = 0; j < 8; ++j) {
        r.u[2 * j]     = f2bf(bp[0]);
        r.u[2 * j + 1] = f2bf(bp[ldb]);
        bp += 2 * ldb;
    }
    return r.bf;
}

// ---------------------------------------------------------------------------
// Kernel P: pack one 32x16 tile per wave into B-fragment order.
//   w: [NE][kTiles*32][ld] f32 row-major, ld = row length (= N dim).
//   ntFast=1 -> tile index = (e*nTiles + nt)*kTiles + kt   (enc: kt fastest)
//   ntFast=0 -> tile index = (e*kTiles + kt)*nTiles + nt   (dec: nt fastest)
// ---------------------------------------------------------------------------
__global__ __launch_bounds__(256) void SwitchSAE_pack(
    const float* __restrict__ w, unsigned short* __restrict__ o,
    int ld, int kTiles, int nTiles, int ntFast) {
    const int lane = threadIdx.x & 31;
    const int tile = blockIdx.x * 8 + (threadIdx.x >> 5);
    if (tile >= NE * kTiles * nTiles) return;
    int kt, nt, e;
    if (ntFast) {           // enc ordering: kt contiguous
        kt = tile % kTiles;
        nt = (tile / kTiles) % nTiles;
        e  = tile / (kTiles * nTiles);
    } else {                // dec ordering: nt contiguous
        nt = tile % nTiles;
        kt = (tile / nTiles) % kTiles;
        e  = tile / (kTiles * nTiles);
    }
    const int g = lane >> 4, hn = lane & 15;
    const float* src = w + (size_t)e * kTiles * 32 * ld
                         + (size_t)(kt * 32 + g * 16) * ld + nt * 16 + hn;
    union { v16u u; v8u h[2]; } r;
#pragma unroll
    for (int j = 0; j < 8; ++j) {
        r.u[2 * j]     = f2bf(src[0]);
        r.u[2 * j + 1] = f2bf(src[ld]);
        src += 2 * (size_t)ld;
    }
    v8u* dst = (v8u*)(o + (size_t)tile * TILE_ELEMS + lane * 16);
    dst[0] = r.h[0];
    dst[1] = r.h[1];
}

// ---------------------------------------------------------------------------
// Kernel 1: router.  One wave per token.
// ---------------------------------------------------------------------------
__global__ __launch_bounds__(256) void SwitchSAE_router(
    const float* __restrict__ act, const float* __restrict__ router_b,
    const float* __restrict__ router, int* __restrict__ counts,
    int* __restrict__ lists, float* __restrict__ scaleArr) {
    const int lane = threadIdx.x & 31;
    const int n = blockIdx.x * 8 + (threadIdx.x >> 5);
    if (n >= NTOK) return;

    float acc[NE];
#pragma unroll
    for (int e = 0; e < NE; ++e) acc[e] = 0.0f;

    const float* xr = act + (size_t)n * DIN;
    for (int d = lane; d < DIN; d += 32) {
        float xv = xr[d] - router_b[d];
        const float* rr = router + (size_t)d * NE;
#pragma unroll
        for (int e = 0; e < NE; ++e) acc[e] += xv * rr[e];
    }
#pragma unroll
    for (int e = 0; e < NE; ++e) {
#pragma unroll
        for (int off = 16; off > 0; off >>= 1)
            acc[e] += __shfl_xor(acc[e], off, 32);
    }
    if (lane == 0) {
        float mx = acc[0];
        int mi = 0;
#pragma unroll
        for (int e = 1; e < NE; ++e)
            if (acc[e] > mx) { mx = acc[e]; mi = e; }
        float s = 0.0f;
#pragma unroll
        for (int e = 0; e < NE; ++e) s += __expf(acc[e] - mx);
        scaleArr[n] = 1.0f / s;                 // max softmax prob
        int pos = atomicAdd(&counts[mi], 1);
        lists[mi * NTOK + pos] = n;
    }
}

// ---------------------------------------------------------------------------
// Kernel 2: grouped fused encode+decode.  256 threads = 8 waves, 32 tokens.
// PACKED=true  -> B operands from pre-packed bf16 fragments (fast path).
// PACKED=false -> B operands converted on the fly from f32 (small-ws fallback).
// ---------------------------------------------------------------------------
template <bool PACKED>
__global__ __launch_bounds__(256) void SwitchSAE_gemm(
    const float* __restrict__ act, const float* __restrict__ b_pre,
    const float* __restrict__ enc, const float* __restrict__ dec,
    const unsigned short* __restrict__ encB, const unsigned short* __restrict__ decB,
    const int* __restrict__ counts, const int* __restrict__ lists,
    const float* __restrict__ scaleArr, float* __restrict__ out) {
    __shared__ __align__(16) unsigned short sXc[MT * PX];   // bf16 Xc tile
    __shared__ __align__(16) unsigned short sL[MT * LP];    // bf16 latent chunk
    __shared__ int   sTok[MT];
    __shared__ int   sVal[MT];
    __shared__ float sScl[MT];

    const int tid  = threadIdx.x;
    const int lane = tid & 31;
    const int wv   = tid >> 5;
    const int bid  = blockIdx.x;

    // --- claim an (expert, tile) slot: scan per-expert tile counts ----------
    int expSel = -1, tile = 0, run = 0;
#pragma unroll
    for (int e = 0; e < NE; ++e) {
        int c = counts[e];
        int t = (c + MT - 1) >> 5;
        if (expSel < 0 && bid < run + t) { expSel = e; tile = bid - run; }
        run += t;
    }
    if (expSel < 0) return;                   // block-uniform early exit
    const int cnt = counts[expSel];

    // --- stage token ids, validity, scales ---------------------------------
    if (tid < MT) {
        int pos = tile * MT + tid;
        int p2 = pos < cnt ? pos : cnt - 1;   // pad rows replicate last token
        int tk = lists[expSel * NTOK + p2];
        sTok[tid] = tk;
        sVal[tid] = (pos < cnt);
        sScl[tid] = scaleArr[tk];
    }
    __syncthreads();

    // --- stage Xc = act[token] - b_pre into LDS as bf16 ---------------------
    for (int i = tid; i < MT * DIN; i += 256) {
        int r = i / DIN;
        int c = i - r * DIN;
        float v = act[(size_t)sTok[r] * DIN + c] - b_pre[c];
        sXc[r * PX + c] = f2bf(v);
    }
    __syncthreads();

    const float* encE = enc + (size_t)expSel * DIN * DK;
    const float* decE = dec + (size_t)expSel * DK * DIN;
    const int g  = lane >> 4;
    const int hn = lane & 15;

    const unsigned short* aRow0 = sXc + (size_t)hn * PX;         // rows 0..15
    const unsigned short* aRow1 = sXc + (size_t)(hn + 16) * PX;  // rows 16..31

    v8f rAcc[2][6];
#pragma unroll
    for (int rb = 0; rb < 2; ++rb)
#pragma unroll
        for (int cs = 0; cs < 6; ++cs) rAcc[rb][cs] = (v8f)0.0f;

    for (int kc = 0; kc < DK; kc += CH) {
        // ---- encode: this wave produces latent cols [kc+wv*16, +16) --------
        v8f eAcc0 = (v8f)0.0f, eAcc1 = (v8f)0.0f;
        {
            // packed: tile index = (e*ENC_NT + nt)*ENC_KT + kt, kt contiguous
            const unsigned short* tp =
                encB + (size_t)((expSel * ENC_NT + (kc >> 4) + wv) * ENC_KT) * TILE_ELEMS;
            const float* bpF = encE + (size_t)(g * 16) * DK + (kc + wv * 16 + hn);
            for (int kk = 0; kk < DIN; kk += 32) {
                v16bf a0 = load_a_lds(aRow0, kk, g);
                v16bf a1 = load_a_lds(aRow1, kk, g);
                v16bf b;
                if constexpr (PACKED) {
                    __builtin_prefetch(tp + TILE_ELEMS, 0, 1);
                    b = load_b_packed(tp, lane);
                    tp += TILE_ELEMS;
                } else {
                    b = load_b_f32(bpF, DK);
                    bpF += (size_t)32 * DK;
                }
                eAcc0 = wmma_bf16(a0, b, eAcc0);
                eAcc1 = wmma_bf16(a1, b, eAcc1);
            }
        }
        // ---- relu + pack into shared latent chunk --------------------------
        {
            const int lcol = wv * 16 + hn;
#pragma unroll
            for (int v = 0; v < 8; ++v) {
                float x0 = eAcc0[v]; x0 = x0 > 0.0f ? x0 : 0.0f;
                float x1 = eAcc1[v]; x1 = x1 > 0.0f ? x1 : 0.0f;
                sL[(v + 8 * g) * LP + lcol]      = f2bf(x0);
                sL[(v + 8 * g + 16) * LP + lcol] = f2bf(x1);
            }
        }
        __syncthreads();

        // ---- decode: consume chunk; wave owns output cols [wv*96, +96) -----
        const unsigned short* lRow0 = sL + (size_t)hn * LP;
        const unsigned short* lRow1 = sL + (size_t)(hn + 16) * LP;
        for (int kk = 0; kk < CH; kk += 32) {
            v16bf a0 = load_a_lds(lRow0, kk, g);
            v16bf a1 = load_a_lds(lRow1, kk, g);
            // packed: tile index = (e*DEC_KT + kt)*DEC_NT + nt, nt contiguous
            const unsigned short* tp =
                decB + (size_t)((expSel * DEC_KT + ((kc + kk) >> 5)) * DEC_NT + wv * 6) * TILE_ELEMS;
            const float* bpF = decE + (size_t)(kc + kk + g * 16) * DIN + wv * 96 + hn;
#pragma unroll
            for (int cs = 0; cs < 6; ++cs) {
                v16bf b;
                if constexpr (PACKED) {
                    b = load_b_packed(tp + (size_t)cs * TILE_ELEMS, lane);
                } else {
                    b = load_b_f32(bpF + cs * 16, DIN);
                }
                rAcc[0][cs] = wmma_bf16(a0, b, rAcc[0][cs]);
                rAcc[1][cs] = wmma_bf16(a1, b, rAcc[1][cs]);
            }
        }
        __syncthreads();   // protect sL before next chunk overwrites it
    }

    // --- epilogue: out[token] = scale*recon + b_pre (valid rows only) -------
#pragma unroll
    for (int rb = 0; rb < 2; ++rb) {
#pragma unroll
        for (int cs = 0; cs < 6; ++cs) {
            const int colg = wv * 96 + cs * 16 + hn;
            const float bpv = b_pre[colg];
#pragma unroll
            for (int v = 0; v < 8; ++v) {
                int m = v + 8 * g + 16 * rb;
                if (sVal[m]) {
                    out[(size_t)sTok[m] * DIN + colg] =
                        sScl[m] * rAcc[rb][cs][v] + bpv;
                }
            }
        }
    }
}

// ---------------------------------------------------------------------------
extern "C" void kernel_launch(void* const* d_in, const int* in_sizes, int n_in,
                              void* d_out, int out_size, void* d_ws, size_t ws_size,
                              hipStream_t stream) {
    const float* act      = (const float*)d_in[0];   // [N, D_IN]
    const float* b_pre    = (const float*)d_in[1];   // [D_IN]
    const float* enc      = (const float*)d_in[2];   // [E, D_IN, K]
    const float* dec      = (const float*)d_in[3];   // [E, K, D_IN]
    const float* router_b = (const float*)d_in[4];   // [D_IN]
    const float* router   = (const float*)d_in[5];   // [D_IN, E]
    float* out = (float*)d_out;

    // ws: counts(64B) | lists(8*N ints) | scale(N f32) | [encB | decB] bf16
    int*   counts = (int*)d_ws;
    int*   lists  = counts + 16;
    float* scl    = (float*)(lists + NE * NTOK);

    const size_t wbase  = 327680;                       // 320 KB header region
    const size_t wElems = (size_t)NE * DIN * DK;        // 18.87M per weight set
    const size_t need   = wbase + 2 * wElems * sizeof(unsigned short); // ~72.3MB
    unsigned short* encB = (unsigned short*)((char*)d_ws + wbase);
    unsigned short* decB = encB + wElems;
    const bool packed = (ws_size >= need);              // deterministic choice

    hipMemsetAsync(counts, 0, 16 * sizeof(int), stream);

    if (packed) {
        // 8*24*192 = 8*96*48 = 36864 tiles each, 8 tiles per block
        SwitchSAE_pack<<<36864 / 8, 256, 0, stream>>>(enc, encB, DK, ENC_KT, ENC_NT, 1);
        SwitchSAE_pack<<<36864 / 8, 256, 0, stream>>>(dec, decB, DIN, DEC_KT, DEC_NT, 0);
    }

    SwitchSAE_router<<<NTOK / 8, 256, 0, stream>>>(act, router_b, router,
                                                   counts, lists, scl);

    if (packed) {
        SwitchSAE_gemm<true><<<NTOK / MT + NE, 256, 0, stream>>>(
            act, b_pre, enc, dec, encB, decB, counts, lists, scl, out);
    } else {
        SwitchSAE_gemm<false><<<NTOK / MT + NE, 256, 0, stream>>>(
            act, b_pre, enc, dec, encB, decB, counts, lists, scl, out);
    }
}
